// DiffPoolEncoder_28681791603046
// MI455X (gfx1250) — compile-verified, matrix operand order checked by reading, not run on hardware
//
#include <hip/hip_runtime.h>

// ---------------------------------------------------------------------------
// DiffPool encoder for MI455X (gfx1250, wave32).
// Adjacency is block-diagonal (64 graphs x 64 nodes) so every matmul becomes
// either a batched small GEMM or a dense GEMM; all run through
// v_wmma_f32_16x16x32_f16 (f16 operands, f32 accumulate). Problem is tiny
// (~3.5 GFLOP, ~36MB working set -> fully L2-resident at 192MB), so the win
// is keeping everything on the matrix pipe with fused epilogues, branchless
// vectorized fragment loads, and fully compile-time-specialized shapes so
// all addressing collapses to immediate offsets inside s_clause'd loads.
// ---------------------------------------------------------------------------

typedef __attribute__((ext_vector_type(16))) _Float16 v16h;
typedef __attribute__((ext_vector_type(8)))  float    v8f;

#define NPG   64
#define BB    64
#define DD    256
#define EE    65536

union HFrag { v16h v; _Float16 h[16]; };
union CFrag { v8f  v; float    f[8];  };
union F4    { float4 q; float f[4];   };

// A fragment: 16x32 tile, rows tile_m..+15, cols k0..+31 of an f32 matrix.
// CDNA5 layout: lanes 0-15 = rows, K in {0..7} U {16..23}; lanes 16-31 get
// K in {8..15} U {24..31}. TRANS_A reads element (m,k) = A[k*LD + m].
// CHECK: branchless — clamp row to 0, select fragment to zero after load.
template<bool TRANS_A, bool CHECK, int LD>
__device__ inline v16h load_a_frag(const float* __restrict__ A,
                                   int tile_m, int k0, int Mbound) {
    int l  = threadIdx.x & 31;
    int m  = tile_m + (l & 15);
    int kb = k0 + ((l >> 4) << 3);
    bool mv = true;
    if (CHECK) { mv = (m < Mbound); if (!mv) m = 0; }
    HFrag fr;
    if (!TRANS_A) {
        const float* row = A + (long)m * LD + kb;   // 16B-aligned: LD%4==0, kb%8==0
        F4 u0, u1, u2, u3;
        u0.q = *(const float4*)(row);
        u1.q = *(const float4*)(row + 4);
        u2.q = *(const float4*)(row + 16);
        u3.q = *(const float4*)(row + 20);
#pragma unroll
        for (int i = 0; i < 4; ++i) {
            fr.h[i]      = (_Float16)u0.f[i];
            fr.h[4 + i]  = (_Float16)u1.f[i];
            fr.h[8 + i]  = (_Float16)u2.f[i];
            fr.h[12 + i] = (_Float16)u3.f[i];
        }
    } else {
        const float* col = A + (long)kb * LD + m;
#pragma unroll
        for (int i = 0; i < 8; ++i) {
            fr.h[i]     = (_Float16)col[i * LD];          // immediate offsets
            fr.h[8 + i] = (_Float16)col[(16 + i) * LD];
        }
    }
    if (CHECK) {
#pragma unroll
        for (int i = 0; i < 16; ++i) fr.h[i] = mv ? fr.h[i] : (_Float16)0.f;
    }
    return fr.v;
}

// B fragment: 32x16 tile (K x N), row-major KxN source; N striped over lanes
// (lane-coalesced loads), one base address + immediate offsets.
template<bool CHECK, int LD>
__device__ inline v16h load_b_frag(const float* __restrict__ B,
                                   int k0, int tile_n, int Nbound) {
    int l  = threadIdx.x & 31;
    int n  = tile_n + (l & 15);
    int kb = k0 + ((l >> 4) << 3);
    bool nv = true;
    if (CHECK) { nv = (n < Nbound); if (!nv) n = 0; }
    const float* p = B + (long)kb * LD + n;
    HFrag fr;
#pragma unroll
    for (int i = 0; i < 8; ++i) {
        fr.h[i]     = (_Float16)p[i * LD];                // immediate offsets
        fr.h[8 + i] = (_Float16)p[(16 + i) * LD];
    }
    if (CHECK) {
#pragma unroll
        for (int i = 0; i < 16; ++i) fr.h[i] = nv ? fr.h[i] : (_Float16)0.f;
    }
    return fr.v;
}

// One wave computes one 16x16 output tile. Batched via compile-time strides;
// optional A-transpose, residual add, bias, ReLU fused into the epilogue.
// Two accumulators break the WMMA->WMMA RAW chain across K iterations.
template<bool TRANS_A, bool RELU, bool RES, bool BIAS, bool CHECK,
         int LDA, int LDB, int LDC, int LDR,
         long SA, long SB, long SC, long SR,
         int M, int N, int K, int TM, int TN, int NB>
__global__ void wmma_gemm_kernel(const float* __restrict__ A,
                                 const float* __restrict__ B,
                                 float* __restrict__ C,
                                 const float* __restrict__ R,
                                 const float* __restrict__ bias) {
    constexpr int nTiles = NB * TM * TN;
    int wave = blockIdx.x * (blockDim.x >> 5) + (threadIdx.x >> 5);
    if (wave >= nTiles) return;                 // wave-uniform: EXEC stays full
    constexpr int perB = TM * TN;
    int b   = wave / perB;
    int rem = wave - b * perB;
    int tm  = (rem / TN) << 4;
    int tn  = (rem % TN) << 4;

    const float* Ab = A + (long)b * SA;
    const float* Bb = B + (long)b * SB;
    float*       Cb = C + (long)b * SC;
    const float* Rb = RES ? (R + (long)b * SR) : nullptr;

    v8f acc0 = {}, acc1 = {};
    int k0 = 0;
#pragma unroll
    for (; k0 + 64 <= K; k0 += 64) {
        v16h a0 = load_a_frag<TRANS_A, CHECK, LDA>(Ab, tm, k0, M);
        v16h b0 = load_b_frag<CHECK, LDB>(Bb, k0, tn, N);
        acc0 = __builtin_amdgcn_wmma_f32_16x16x32_f16(
            false, a0, false, b0, (short)0, acc0, false, false);
        v16h a1 = load_a_frag<TRANS_A, CHECK, LDA>(Ab, tm, k0 + 32, M);
        v16h b1 = load_b_frag<CHECK, LDB>(Bb, k0 + 32, tn, N);
        acc1 = __builtin_amdgcn_wmma_f32_16x16x32_f16(
            false, a1, false, b1, (short)0, acc1, false, false);
    }
    if (k0 < K) {   // K == 32 tail
        v16h a0 = load_a_frag<TRANS_A, CHECK, LDA>(Ab, tm, k0, M);
        v16h b0 = load_b_frag<CHECK, LDB>(Bb, k0, tn, N);
        acc0 = __builtin_amdgcn_wmma_f32_16x16x32_f16(
            false, a0, false, b0, (short)0, acc0, false, false);
    }

    CFrag cf0, cf1;
    cf0.v = acc0; cf1.v = acc1;
    int l  = threadIdx.x & 31;
    int n  = tn + (l & 15);
    int mb = tm + ((l >> 4) << 3);
    const float* rp = RES ? (Rb + (long)mb * LDR + n) : nullptr;
    float*       cp = Cb + (long)mb * LDC + n;
#pragma unroll
    for (int i = 0; i < 8; ++i) {
        float v = cf0.f[i] + cf1.f[i];
        if (BIAS) v += bias[n];
        if (RES)  v += rp[i * LDR];
        if (RELU) v = v > 0.f ? v : 0.f;
        if (CHECK) {
            if (mb + i < M && n < N) cp[i * LDC] = v;
        } else {
            cp[i * LDC] = v;
        }
    }
}

__global__ void zero_kernel(float* p, int n) {
    int i = blockIdx.x * blockDim.x + threadIdx.x;
    if (i < n) p[i] = 0.f;
}

// A[src,dst] = 1.0 ; adjacency stored as 64 blocks of 64x64
__global__ void build_adj_kernel(const int* __restrict__ src,
                                 const int* __restrict__ dst,
                                 float* __restrict__ Ab) {
    int e = blockIdx.x * blockDim.x + threadIdx.x;
    if (e < EE) {
        int s = src[e], d = dst[e];
        int g = s >> 6;
        Ab[(g << 12) + ((s & 63) << 6) + (d & 63)] = 1.0f;
    }
}

// row-wise softmax, C <= 32, one thread per row (rows are tiny, L2-resident)
__global__ void softmax_rows_kernel(float* __restrict__ X, int Rn, int Cn) {
    int r = blockIdx.x * blockDim.x + threadIdx.x;
    if (r >= Rn) return;
    float* row = X + (long)r * Cn;
    float mx = row[0];
    for (int c = 1; c < Cn; ++c) mx = fmaxf(mx, row[c]);
    float s = 0.f;
    for (int c = 0; c < Cn; ++c) { float e = __expf(row[c] - mx); row[c] = e; s += e; }
    float inv = 1.f / s;
    for (int c = 0; c < Cn; ++c) row[c] *= inv;
}

// Xg[b,d] = mean over P2 coarse nodes
__global__ void mean_rows_kernel(const float* __restrict__ X2, float* __restrict__ Xg) {
    int i = blockIdx.x * blockDim.x + threadIdx.x;
    if (i >= BB * DD) return;
    int g = i >> 8, d = i & 255;
    float s = 0.f;
#pragma unroll
    for (int p = 0; p < 8; ++p) s += X2[(long)((g << 3) + p) * DD + d];
    Xg[i] = s * 0.125f;
}

// All shape/stride parameters are compile-time template arguments.
#define LG(TA, RL, RS, BI, CK, LDA, LDB, LDC, LDR, SA, SB, SC, SR, M, N, K, TM, TN, NB, Ap, Bp, Cp, Rp, bp) \
    wmma_gemm_kernel<TA, RL, RS, BI, CK, LDA, LDB, LDC, LDR, SA, SB, SC, SR, M, N, K, TM, TN, NB>           \
        <<<((NB) * (TM) * (TN) + 3) / 4, 128, 0, stream>>>(Ap, Bp, Cp, Rp, bp)

extern "C" void kernel_launch(void* const* d_in, const int* in_sizes, int n_in,
                              void* d_out, int out_size, void* d_ws, size_t ws_size,
                              hipStream_t stream) {
    const float* x        = (const float*)d_in[0];
    const int*   ei       = (const int*)d_in[1];
    const int*   src      = ei;
    const int*   dst      = ei + EE;
    const float* pool0_W1 = (const float*)d_in[3];
    const float* pool0_W2 = (const float*)d_in[4];
    const float* emb0_W1  = (const float*)d_in[5];
    const float* emb0_W2  = (const float*)d_in[6];
    const float* pool1_W1 = (const float*)d_in[7];
    const float* pool1_W2 = (const float*)d_in[8];
    const float* emb1_W1  = (const float*)d_in[9];
    const float* emb1_W2  = (const float*)d_in[10];
    const float* lin1_W   = (const float*)d_in[11];
    const float* lin1_b   = (const float*)d_in[12];
    const float* lin2_W   = (const float*)d_in[13];
    const float* lin2_b   = (const float*)d_in[14];
    float*       out      = (float*)d_out;
    const float* nul      = nullptr;

    // workspace layout (floats); level-2 buffers overlay dead level-1 buffers
    float* ws = (float*)d_ws;
    float* Ab = ws + 0;             // 64 x (64x64)        262144
    float* T  = ws + 262144;        // 4096x256           1048576
    float* Hp = ws + 1310720;       // 4096x256
    float* He = ws + 2359296;       // 4096x256
    float* Up = ws + 3407872;       // 4096x256
    float* Ue = ws + 4456448;       // 4096x256
    float* S1 = ws + 5505024;       // 4096x32             131072
    float* Z1 = ws + 5636096;       // 4096x256
    float* X1 = ws + 6684672;       // 2048x256            524288
    float* Mb = ws + 7208960;       // 64 x (64x32)        131072
    float* A1 = ws + 7340032;       // 64 x (32x32)         65536
    float* T2 = T;  float* Hp2 = Hp; float* He2 = He;      // overlays
    float* Up2 = Up; float* Ue2 = Ue;
    float* S2 = S1; float* Z2 = Z1; float* X2 = Mb;
    float* Xg = ws + 7405568;       // 64x256               16384
    float* L1 = ws + 7421952;       // 64x512               32768
    (void)in_sizes; (void)n_in; (void)out_size; (void)ws_size;

    // ---- adjacency: zero + scatter (block-diagonal, 64 blocks of 64x64) ----
    zero_kernel<<<(BB * NPG * NPG + 255) / 256, 256, 0, stream>>>(Ab, BB * NPG * NPG);
    build_adj_kernel<<<(EE + 255) / 256, 256, 0, stream>>>(src, dst, Ab);

    // ================= level 1 (nodes = 4096, block = 64) =================
    // T = A@x + x (batched block-diag spmm as dense 64x64 GEMM + residual)
    LG(false, false, true, false, false, 64, 256, 256, 256, 4096, 16384, 16384, 16384,
       64, 256, 64, 4, 16, BB, Ab, x, T, x, nul);
    // Hp = relu(T @ pool0_W1) ; He = relu(T @ emb0_W1)
    LG(false, true, false, false, false, 256, 256, 256, 0, 0, 0, 0, 0,
       4096, 256, 256, 256, 16, 1, T, pool0_W1, Hp, nul, nul);
    LG(false, true, false, false, false, 256, 256, 256, 0, 0, 0, 0, 0,
       4096, 256, 256, 256, 16, 1, T, emb0_W1, He, nul, nul);
    // Up = A@Hp + Hp ; Ue = A@He + He
    LG(false, false, true, false, false, 64, 256, 256, 256, 4096, 16384, 16384, 16384,
       64, 256, 64, 4, 16, BB, Ab, Hp, Up, Hp, nul);
    LG(false, false, true, false, false, 64, 256, 256, 256, 4096, 16384, 16384, 16384,
       64, 256, 64, 4, 16, BB, Ab, He, Ue, He, nul);
    // S1 = Up @ pool0_W2 (4096x32) ; Z1 = Ue @ emb0_W2 (4096x256)
    LG(false, false, false, false, false, 256, 32, 32, 0, 0, 0, 0, 0,
       4096, 32, 256, 256, 2, 1, Up, pool0_W2, S1, nul, nul);
    LG(false, false, false, false, false, 256, 256, 256, 0, 0, 0, 0, 0,
       4096, 256, 256, 256, 16, 1, Ue, emb0_W2, Z1, nul, nul);
    softmax_rows_kernel<<<(4096 + 127) / 128, 128, 0, stream>>>(S1, 4096, 32);
    // X1[b] = S1[b]^T @ Z1[b]   (32x64 @ 64x256)
    LG(true, false, false, false, false, 32, 256, 256, 0, 2048, 16384, 8192, 0,
       32, 256, 64, 2, 16, BB, S1, Z1, X1, nul, nul);
    // Mb[b] = Ab[b] @ S1[b] ; A1[b] = S1[b]^T @ Mb[b]
    LG(false, false, false, false, false, 64, 32, 32, 0, 4096, 2048, 2048, 0,
       64, 32, 64, 4, 2, BB, Ab, S1, Mb, nul, nul);
    LG(true, false, false, false, false, 32, 32, 32, 0, 2048, 2048, 1024, 0,
       32, 32, 64, 2, 2, BB, S1, Mb, A1, nul, nul);

    // ================= level 2 (nodes = 2048, block = 32) =================
    LG(false, false, true, false, false, 32, 256, 256, 256, 1024, 8192, 8192, 8192,
       32, 256, 32, 2, 16, BB, A1, X1, T2, X1, nul);
    LG(false, true, false, false, false, 256, 256, 256, 0, 0, 0, 0, 0,
       2048, 256, 256, 128, 16, 1, T2, pool1_W1, Hp2, nul, nul);
    LG(false, true, false, false, false, 256, 256, 256, 0, 0, 0, 0, 0,
       2048, 256, 256, 128, 16, 1, T2, emb1_W1, He2, nul, nul);
    LG(false, false, true, false, false, 32, 256, 256, 256, 1024, 8192, 8192, 8192,
       32, 256, 32, 2, 16, BB, A1, Hp2, Up2, Hp2, nul);
    LG(false, false, true, false, false, 32, 256, 256, 256, 1024, 8192, 8192, 8192,
       32, 256, 32, 2, 16, BB, A1, He2, Ue2, He2, nul);
    // S2 = Up2 @ pool1_W2 (2048x8, N<16 handled by branchless CHECK path)
    LG(false, false, false, false, true, 256, 8, 8, 0, 0, 0, 0, 0,
       2048, 8, 256, 128, 1, 1, Up2, pool1_W2, S2, nul, nul);
    LG(false, false, false, false, false, 256, 256, 256, 0, 0, 0, 0, 0,
       2048, 256, 256, 128, 16, 1, Ue2, emb1_W2, Z2, nul, nul);
    softmax_rows_kernel<<<(2048 + 127) / 128, 128, 0, stream>>>(S2, 2048, 8);
    // X2[b] = S2[b]^T @ Z2[b]   (8x32 @ 32x256), M=8 via branchless CHECK
    LG(true, false, false, false, true, 8, 256, 256, 0, 256, 8192, 2048, 0,
       8, 256, 32, 1, 16, BB, S2, Z2, X2, nul, nul);

    // ====================== readout + MLP head ============================
    mean_rows_kernel<<<(BB * DD + 255) / 256, 256, 0, stream>>>(X2, Xg);
    LG(false, true, false, true, false, 256, 512, 512, 0, 0, 0, 0, 0,
       64, 512, 256, 4, 32, 1, Xg, lin1_W, L1, nul, lin1_b);
    LG(false, false, false, true, false, 512, 256, 256, 0, 0, 0, 0, 0,
       64, 256, 512, 4, 16, 1, L1, lin2_W, out, nul, lin2_b);
}
